// Codebook_74259984547920
// MI455X (gfx1250) — matrix operand order repaired and checked by reading, not compiled, not run on hardware
//
#include <hip/hip_runtime.h>

typedef __attribute__((ext_vector_type(16))) __bf16 v16bf;
typedef __attribute__((ext_vector_type(8)))  float  v8f;

#define DDIM 256
#define KDIM 1024

// ---------------- prep: c2 = row norms of C ----------------
__global__ void c2_kernel(const float* __restrict__ C, float* __restrict__ c2) {
    int k = blockIdx.x * blockDim.x + threadIdx.x;
    if (k < KDIM) {
        const float* row = C + (size_t)k * DDIM;
        float s = 0.f;
        for (int j = 0; j < DDIM; ++j) s = fmaf(row[j], row[j], s);
        c2[k] = s;
    }
}

// ---------------- prep: split C into bf16 hi/lo, pre-swizzled into the
// 32x16 B lane layout used here (contiguous k order, matching A): flat index [ntile][kk][lane][elem].
// lane<16 holds K = kk*32 + elem (0..15), lane>=16 holds K = kk*32+16+elem.
// column N = ntile*16 + lane%16.  B(K,N) = C[N][K]  (B = C^T).
__global__ void prep_swizzle(const float* __restrict__ C,
                             __bf16* __restrict__ Bhi, __bf16* __restrict__ Blo) {
    int g = blockIdx.x * blockDim.x + threadIdx.x;   // 0 .. 64*8*32*16-1
    int e    = g & 15;
    int lane = (g >> 4) & 31;
    int kk   = (g >> 9) & 7;
    int nb   = g >> 12;
    int col  = nb * 16 + (lane & 15);
    int K    = kk * 32 + ((lane & 16) ? 16 : 0) + e;
    float v  = C[(size_t)col * DDIM + K];
    __bf16 h = (__bf16)v;
    Bhi[g] = h;
    Blo[g] = (__bf16)(v - (float)h);
}

// ---------------- main fused kernel ----------------
__global__ __launch_bounds__(256) void
codebook_kernel(const float* __restrict__ Q, const float* __restrict__ c2,
                const __bf16* __restrict__ Bhi, const __bf16* __restrict__ Blo,
                float* __restrict__ out) {
    __shared__ float q2part[256];
    __shared__ float q2s[16];
    __shared__ float red_f[16 * 128];
    __shared__ int   red_i[16 * 128];

    const int tid  = threadIdx.x;
    const int wave = tid >> 5;        // 0..7  (wave32)
    const int lane = tid & 31;
    const int l16  = lane & 15;
    const int lhi  = lane >> 4;       // 0 or 1
    const long m0  = (long)blockIdx.x * 16;   // 16 Q-rows per workgroup

    // ---- q2 for our 16 rows ----
    {
        int row = tid >> 4, j0 = tid & 15;
        const float* qr = Q + (m0 + row) * DDIM;
        float s = 0.f;
        for (int j = j0; j < DDIM; j += 16) { float v = qr[j]; s = fmaf(v, v, s); }
        q2part[tid] = s;
    }
    __syncthreads();
    if (tid < 16) {
        float s = 0.f;
        for (int j = 0; j < 16; ++j) s += q2part[tid * 16 + j];
        q2s[tid] = s;
    }
    __syncthreads();

    // ---- GEMM: acc[n] = Q_tile . C_tile^T  via bf16x3 split WMMA ----
    v8f acc[8] = {};
    const float* qrow = Q + (m0 + l16) * DDIM;   // A row = lane%16

    // Per-wave/lane base into the swizzled B arrays; every tile load below is
    // base + compile-time offset (fits the 24-bit IOFFSET -> no per-iter addr VALU).
    const v16bf* bh_base = (const v16bf*)(Bhi + (size_t)wave * 32768 + (size_t)lane * 16);
    const v16bf* bl_base = (const v16bf*)(Blo + (size_t)wave * 32768 + (size_t)lane * 16);

    for (int kk = 0; kk < 8; ++kk) {
        // A 16x32 bf16, contiguous k order (same as B): lane<16 -> K {kk*32+0..15}; lane>=16 -> {+16..31}
        int k0 = kk * 32 + (lhi ? 16 : 0);
        float4 f0 = *(const float4*)(qrow + k0);
        float4 f1 = *(const float4*)(qrow + k0 + 4);
        float4 f2 = *(const float4*)(qrow + k0 + 8);
        float4 f3 = *(const float4*)(qrow + k0 + 12);
        float fa[16] = { f0.x, f0.y, f0.z, f0.w, f1.x, f1.y, f1.z, f1.w,
                         f2.x, f2.y, f2.z, f2.w, f3.x, f3.y, f3.z, f3.w };
        v16bf ah, al;
#pragma unroll
        for (int i = 0; i < 16; ++i) {
            __bf16 h = (__bf16)fa[i];
            ah[i] = h;
            al[i] = (__bf16)(fa[i] - (float)h);
        }
        // Pair-interleave N-tiles: consecutive WMMAs hit different accumulators,
        // and no VALU touches a just-issued WMMA's source regs (kills hazard nops).
#pragma unroll
        for (int np = 0; np < 4; ++np) {
            const int n0 = 2 * np, n1 = 2 * np + 1;
            v16bf bh0 = bh_base[n0 * 256 + kk * 32];
            v16bf bl0 = bl_base[n0 * 256 + kk * 32];
            v16bf bh1 = bh_base[n1 * 256 + kk * 32];
            v16bf bl1 = bl_base[n1 * 256 + kk * 32];
            acc[n0] = __builtin_amdgcn_wmma_f32_16x16x32_bf16(
                false, ah, false, bh0, (short)0, acc[n0], false, false);
            acc[n1] = __builtin_amdgcn_wmma_f32_16x16x32_bf16(
                false, ah, false, bh1, (short)0, acc[n1], false, false);
            acc[n0] = __builtin_amdgcn_wmma_f32_16x16x32_bf16(
                false, ah, false, bl0, (short)0, acc[n0], false, false);
            acc[n1] = __builtin_amdgcn_wmma_f32_16x16x32_bf16(
                false, ah, false, bl1, (short)0, acc[n1], false, false);
            acc[n0] = __builtin_amdgcn_wmma_f32_16x16x32_bf16(
                false, al, false, bh0, (short)0, acc[n0], false, false);
            acc[n1] = __builtin_amdgcn_wmma_f32_16x16x32_bf16(
                false, al, false, bh1, (short)0, acc[n1], false, false);
        }
    }

    // ---- dist = max(q2 + c2 - 2*S, 0); per-lane min/argmin ----
    float q2r[8];
#pragma unroll
    for (int r = 0; r < 8; ++r) q2r[r] = q2s[r + 8 * lhi];

    float lminv[8]; int lmini[8];
#pragma unroll
    for (int r = 0; r < 8; ++r) { lminv[r] = 3.4e38f; lmini[r] = 0; }

#pragma unroll
    for (int n = 0; n < 8; ++n) {
        int k = wave * 128 + n * 16 + l16;
        float c2v = c2[k];
#pragma unroll
        for (int r = 0; r < 8; ++r) {
            float d = fmaxf(fmaf(-2.f, acc[n][r], q2r[r] + c2v), 0.f);
            acc[n][r] = d;
            if (d < lminv[r]) { lminv[r] = d; lmini[r] = k; }  // k ascending -> first-index ties
        }
    }

    // ---- workgroup min/argmin reduce (per row, 128 slots) ----
#pragma unroll
    for (int r = 0; r < 8; ++r) {
        int row = r + 8 * lhi, slot = wave * 16 + l16;
        red_f[row * 128 + slot] = lminv[r];
        red_i[row * 128 + slot] = lmini[r];
    }
    __syncthreads();
    for (int s = 64; s > 0; s >>= 1) {
        for (int p = tid; p < 16 * s; p += 256) {
            int row = p / s, i = p - row * s;
            int ia = row * 128 + i, ib = ia + s;
            float av = red_f[ia], bv = red_f[ib];
            int   ai = red_i[ia], bi = red_i[ib];
            if (bv < av || (bv == av && bi < ai)) { red_f[ia] = bv; red_i[ia] = bi; }
        }
        __syncthreads();
    }
    float dminr[8]; int kstar[8];
#pragma unroll
    for (int r = 0; r < 8; ++r) {
        int row = r + 8 * lhi;
        dminr[r] = red_f[row * 128];
        kstar[r] = red_i[row * 128];
    }
    __syncthreads();   // before reusing red_f

    // ---- softmax numerator: exp((dmin - d)/TAU) = exp2((dmin-d)*2/ln2) ----
    const float K2 = 2.8853900817779268f;   // 2 / ln(2)
    float psum[8];
#pragma unroll
    for (int r = 0; r < 8; ++r) psum[r] = 0.f;
#pragma unroll
    for (int n = 0; n < 8; ++n)
#pragma unroll
        for (int r = 0; r < 8; ++r) {
            float e = exp2f((dminr[r] - acc[n][r]) * K2);
            acc[n][r] = e;
            psum[r] += e;
        }
#pragma unroll
    for (int r = 0; r < 8; ++r)
        red_f[(r + 8 * lhi) * 128 + wave * 16 + l16] = psum[r];
    __syncthreads();
    for (int s = 64; s > 0; s >>= 1) {
        for (int p = tid; p < 16 * s; p += 256) {
            int row = p / s, i = p - row * s;
            red_f[row * 128 + i] += red_f[row * 128 + i + s];
        }
        __syncthreads();
    }
    float rsum[8];
#pragma unroll
    for (int r = 0; r < 8; ++r) rsum[r] = 1.0f / red_f[(r + 8 * lhi) * 128];

    // ---- out = (one_hot - softmax) + softmax ; NT stores (write-once 512MB) ----
#pragma unroll
    for (int n = 0; n < 8; ++n) {
        int k = wave * 128 + n * 16 + l16;
#pragma unroll
        for (int r = 0; r < 8; ++r) {
            int row = r + 8 * lhi;
            float s = acc[n][r] * rsum[r];
            float h = (k == kstar[r]) ? 1.0f : 0.0f;
            __builtin_nontemporal_store((h - s) + s, &out[(m0 + row) * KDIM + k]);
        }
    }
}

extern "C" void kernel_launch(void* const* d_in, const int* in_sizes, int n_in,
                              void* d_out, int out_size, void* d_ws, size_t ws_size,
                              hipStream_t stream) {
    const float* Q = reinterpret_cast<const float*>(d_in[0]);   // [131072, 256]
    const float* C = reinterpret_cast<const float*>(d_in[1]);   // [1024, 256]
    float* out = reinterpret_cast<float*>(d_out);               // [131072, 1024]

    float*  c2  = reinterpret_cast<float*>(d_ws);                         // 4 KB
    __bf16* Bhi = reinterpret_cast<__bf16*>((char*)d_ws + 4096);          // 512 KB
    __bf16* Blo = Bhi + (size_t)KDIM * DDIM;                              // 512 KB

    c2_kernel<<<dim3(4), dim3(256), 0, stream>>>(C, c2);
    prep_swizzle<<<dim3((KDIM * DDIM) / 256), dim3(256), 0, stream>>>(C, Bhi, Blo);
    codebook_kernel<<<dim3(131072 / 16), dim3(256), 0, stream>>>(Q, c2, Bhi, Blo, out);
}
